// HQNN_45509473468977
// MI455X (gfx1250) — matrix operand order repaired and checked by reading, not compiled, not run on hardware
//
#include <hip/hip_runtime.h>

typedef __attribute__((ext_vector_type(16))) __bf16 v16bf;
typedef __attribute__((ext_vector_type(8)))  float  v8f;

union ABf { uint4 u[2]; v16bf v; };

__device__ __forceinline__ unsigned short f2bf(float f) {
    unsigned int u = __float_as_uint(f);
    unsigned int r = u + 0x7FFFu + ((u >> 16) & 1u);
    return (unsigned short)(r >> 16);
}
__device__ __forceinline__ float bf2f(unsigned short h) {
    return __uint_as_float(((unsigned int)h) << 16);
}

// ---------------------------------------------------------------------------
// Kernel 1: conv1 3x3 VALID (1->32 ch) + bias + ReLU, write bf16 NHWC
// h1[((n*26+y)*26+x)*32 + c]
// ---------------------------------------------------------------------------
__global__ __launch_bounds__(256) void conv1_kernel(
    const float* __restrict__ x, const float* __restrict__ w,
    const float* __restrict__ b, unsigned short* __restrict__ h1)
{
    __shared__ float sw[288];
    __shared__ float sb[32];
    int tid = threadIdx.x;
    for (int i = tid; i < 288; i += 256) sw[i] = w[i];
    if (tid < 32) sb[tid] = b[tid];
    __syncthreads();

    long gid = (long)blockIdx.x * 256 + tid;      // < 2048*26*26 = 1,384,448
    int xx = (int)(gid % 26);
    long r  = gid / 26;
    int yy = (int)(r % 26);
    int n  = (int)(r / 26);

    float vin[9];
#pragma unroll
    for (int dy = 0; dy < 3; dy++)
#pragma unroll
        for (int dx = 0; dx < 3; dx++)
            vin[dy * 3 + dx] = x[(long)n * 784 + (yy + dy) * 28 + (xx + dx)];

    unsigned int pack[16];
#pragma unroll
    for (int c = 0; c < 32; c++) {
        float acc = sb[c];
#pragma unroll
        for (int k = 0; k < 9; k++) acc = fmaf(sw[c * 9 + k], vin[k], acc);
        acc = fmaxf(acc, 0.0f);
        unsigned short hv = f2bf(acc);
        if (c & 1) pack[c >> 1] |= ((unsigned int)hv) << 16;
        else       pack[c >> 1]  = (unsigned int)hv;
    }
    uint4* dst = (uint4*)(h1 + gid * 32);
#pragma unroll
    for (int i = 0; i < 4; i++)
        dst[i] = make_uint4(pack[4 * i], pack[4 * i + 1], pack[4 * i + 2], pack[4 * i + 3]);
}

// ---------------------------------------------------------------------------
// Kernel 2: fc1 weight f32 -> bf16, WITH K-permutation NCHW->NHWC:
// wb[j][s*64 + c] = wf[j][c*144 + s]   (s = y2*12+x2, 9216 = 144*64)
// This makes B's K-order match the NHWC pooled activations, so the pool
// kernel can write fully coalesced.
// ---------------------------------------------------------------------------
__global__ __launch_bounds__(256) void convert_w_kernel(
    const float* __restrict__ wf, unsigned short* __restrict__ wb)
{
    int i = blockIdx.x * 256 + threadIdx.x;   // exactly 32*9216 = 294912
    int j  = i / 9216;
    int k2 = i - j * 9216;
    int s  = k2 >> 6;
    int c  = k2 & 63;
    wb[i] = f2bf(wf[j * 9216 + c * 144 + s]);
}

// ---------------------------------------------------------------------------
// Kernel 3: conv2 as implicit GEMM via v_wmma_f32_16x16x32_bf16
// M = 2048*24*24 rows (im2col), N = 64, K = 288 (9 taps x 32 ch)
// block = 8 waves; wave = 16 rows x 64 cols; +bias +ReLU -> bf16 NHWC c2out
// A fragments: per-lane base pointer + immediate tap offsets (no per-ks math)
// B fragments: all 4 hoisted before the 4 WMMAs (one ds wait per k-step)
// ---------------------------------------------------------------------------
__global__ __launch_bounds__(256) void conv2_wmma_kernel(
    const unsigned short* __restrict__ h1, const float* __restrict__ w2,
    const float* __restrict__ b2, unsigned short* __restrict__ c2out)
{
    __shared__ unsigned short Wt[64 * 288];   // [n][k], k = tap*32 + in_ch

    int tid = threadIdx.x;
    for (int i = tid; i < 64 * 288; i += 256) {
        int n = i / 288, k = i % 288;
        int ic = k & 31, tap = k >> 5;        // tap = dy*3+dx
        Wt[n * 288 + k] = f2bf(w2[n * 288 + ic * 9 + tap]);
    }
    __syncthreads();

    int wave = tid >> 5, lane = tid & 31;
    int rowBase = blockIdx.x * 128 + wave * 16;
    int p   = rowBase + (lane & 15);          // this lane's A row
    int n_i = p / 576;
    int rem = p % 576;
    int y   = rem / 24, xo = rem % 24;

    const int choffA = (lane < 16) ? 0 : 16;  // bf16 element offset of 1st K-chunk
    const int koffB  = (lane < 16) ? 0 : 16;

    // per-lane base of the 3x3 patch window (NHWC); taps are constant offsets
    const unsigned short* abase =
        h1 + ((long)(n_i * 26 + y) * 26 + xo) * 32 + choffA;
    const unsigned short* bbase = Wt + (lane & 15) * 288 + koffB;

    v8f acc[4];
#pragma unroll
    for (int nt = 0; nt < 4; nt++) acc[nt] = (v8f)(0.0f);

#pragma unroll
    for (int ks = 0; ks < 9; ks++) {
        const int dy = ks / 3, dx = ks % 3;
        const uint4* ap = (const uint4*)(abase + (dy * 26 + dx) * 32);
        ABf a;
        a.u[0] = ap[0];                        // K chunk [0..8)  (or [16..24))
        a.u[1] = ap[1];                        // K chunk [8..16) (or [24..32))
        ABf bb[4];
#pragma unroll
        for (int nt = 0; nt < 4; nt++) {
            const uint4* bp = (const uint4*)(bbase + nt * (16 * 288) + ks * 32);
            bb[nt].u[0] = bp[0];
            bb[nt].u[1] = bp[1];
        }
#pragma unroll
        for (int nt = 0; nt < 4; nt++)
            acc[nt] = __builtin_amdgcn_wmma_f32_16x16x32_bf16(
                false, a.v, false, bb[nt].v, (short)0, acc[nt], false, false);
    }

#pragma unroll
    for (int nt = 0; nt < 4; nt++) {
        int c = (lane & 15) + nt * 16;
        float bias = b2[c];
#pragma unroll
        for (int v = 0; v < 8; v++) {
            int mm = v + 8 * (lane >> 4);
            long pp = rowBase + mm;
            float val = fmaxf(acc[nt][v] + bias, 0.0f);
            c2out[pp * 64 + c] = f2bf(val);
        }
    }
}

// ---------------------------------------------------------------------------
// Kernel 4: 2x2 maxpool, NHWC -> NHWC (k' = (y2*12+x2)*64 + c).
// Output index == global thread id: fully coalesced reads AND writes.
// ---------------------------------------------------------------------------
__global__ __launch_bounds__(256) void pool_kernel(
    const unsigned short* __restrict__ c2o, unsigned short* __restrict__ pooled)
{
    long t = (long)blockIdx.x * 256 + threadIdx.x;  // 2048*9216
    int c  = (int)(t & 63);
    long r = t >> 6;
    int x2 = (int)(r % 12); r /= 12;
    int y2 = (int)(r % 12); r /= 12;
    int n  = (int)r;
    long base = ((long)(n * 24 + 2 * y2) * 24 + 2 * x2) * 64 + c;
    float m0 = bf2f(c2o[base]);
    float m1 = bf2f(c2o[base + 64]);
    float m2 = bf2f(c2o[base + 24 * 64]);
    float m3 = bf2f(c2o[base + 24 * 64 + 64]);
    pooled[t] = f2bf(fmaxf(fmaxf(m0, m1), fmaxf(m2, m3)));
}

// ---------------------------------------------------------------------------
// Kernel 5: FC (2048x9216 @ 9216x32^T) via WMMA bf16 + bias + ReLU -> f32.
// 64 blocks; block = 32 batch rows. 8 waves = 2 M-subtiles x 4 K-splits
// (72 k-steps each); partial sums reduced through padded LDS.
// ---------------------------------------------------------------------------
__global__ __launch_bounds__(256) void fc_wmma_kernel(
    const unsigned short* __restrict__ pooled, const unsigned short* __restrict__ wfc,
    const float* __restrict__ fcb, float* __restrict__ fcout)
{
    __shared__ float red[8][32][17];   // [wave][lane][nt*8+v], padded vs banks

    int tid = threadIdx.x, wave = tid >> 5, lane = tid & 31;
    int mt  = wave & 1;                // M-subtile within block
    int ksp = wave >> 1;               // K-split id 0..3
    int mbase = blockIdx.x * 32 + mt * 16;
    const unsigned short* arow = pooled + (long)(mbase + (lane & 15)) * 9216;

    const int choffA = (lane < 16) ? 0 : 16;
    const int koffB  = (lane < 16) ? 0 : 16;
    const unsigned short* b0row = wfc + (long)(lane & 15) * 9216 + koffB;

    v8f acc[2];
    acc[0] = (v8f)(0.0f);
    acc[1] = (v8f)(0.0f);

    const int kbeg = ksp * 72;
#pragma unroll 4
    for (int ks = kbeg; ks < kbeg + 72; ks++) {
        ABf a;
        const uint4* ap = (const uint4*)(arow + ks * 32 + choffA);
        a.u[0] = ap[0];
        a.u[1] = ap[1];
        ABf bb[2];
#pragma unroll
        for (int nt = 0; nt < 2; nt++) {
            const uint4* bp = (const uint4*)(b0row + nt * (16 * 9216) + ks * 32);
            bb[nt].u[0] = bp[0];
            bb[nt].u[1] = bp[1];
        }
#pragma unroll
        for (int nt = 0; nt < 2; nt++)
            acc[nt] = __builtin_amdgcn_wmma_f32_16x16x32_bf16(
                false, a.v, false, bb[nt].v, (short)0, acc[nt], false, false);
    }

#pragma unroll
    for (int nt = 0; nt < 2; nt++)
#pragma unroll
        for (int v = 0; v < 8; v++)
            red[wave][lane][nt * 8 + v] = acc[nt][v];
    __syncthreads();

    if (wave < 2) {                    // one reducer wave per M-subtile (mt==wave)
#pragma unroll
        for (int nt = 0; nt < 2; nt++) {
            int c = (lane & 15) + nt * 16;
            float bias = fcb[c];
#pragma unroll
            for (int v = 0; v < 8; v++) {
                int i = nt * 8 + v;
                float s = red[wave][lane][i] + red[wave + 2][lane][i]
                        + red[wave + 4][lane][i] + red[wave + 6][lane][i];
                int mm = v + 8 * (lane >> 4);
                fcout[(long)(blockIdx.x * 32 + wave * 16 + mm) * 32 + c] =
                    fmaxf(s + bias, 0.0f);
            }
        }
    }
}

// ---------------------------------------------------------------------------
// Kernel 6: 5-qubit circuit (32 amplitudes = 32 lanes of one wave) + Z-expect
// + *10 + log_softmax. One wave per batch element.
// ---------------------------------------------------------------------------
__global__ __launch_bounds__(256) void quantum_kernel(
    const float* __restrict__ fcout, float* __restrict__ out)
{
    long gtid = (long)blockIdx.x * 256 + threadIdx.x;
    int b = (int)(gtid >> 5);
    int j = (int)(gtid & 31);
    if (b >= 2048) return;

    float v   = fcout[b * 32 + j];
    float ang = (1.0f / (1.0f + __expf(-v))) * 3.14159265358979f * 0.5f;
    float cl  = __cosf(ang);
    float sl  = __sinf(ang);

    // init state: (-1)^popc(j) / sqrt(32)
    float st = (__popc(j) & 1) ? -0.176776695296637f : 0.176776695296637f;

#pragma unroll
    for (int i = 0; i < 32; i++) {
        const int w = i % 5, t = (i + 2) % 5;
        const int bw = 1 << (4 - w), bt = 1 << (4 - t);
        float ci = __shfl(cl, i, 32);
        float si = __shfl(sl, i, 32);
        // RY on wire w: mix lane j with partner j^bw
        float other = __shfl(st, j ^ bw, 32);
        int bit = (j >> (4 - w)) & 1;
        float ns = bit ? fmaf(si, other, ci * st) : fmaf(ci, st, -si * other);
        // CNOT(w, t): new[j] = old[j ^ bt] if ctrl bit (wire w) of j set
        int src = bit ? (j ^ bt) : j;
        st = __shfl(ns, src, 32);
    }

    float p  = st * st;
    float r0 = p * (1.0f - 2.0f * (float)((j >> 4) & 1));
    float r1 = p * (1.0f - 2.0f * (float)((j >> 3) & 1));
#pragma unroll
    for (int off = 16; off > 0; off >>= 1) {
        r0 += __shfl_xor(r0, off, 32);
        r1 += __shfl_xor(r1, off, 32);
    }
    if (j == 0) {
        float q0 = r0 * 10.0f, q1 = r1 * 10.0f;
        float m = fmaxf(q0, q1);
        float lse = m + __logf(__expf(q0 - m) + __expf(q1 - m));
        out[b * 2 + 0] = q0 - lse;
        out[b * 2 + 1] = q1 - lse;
    }
}

// ---------------------------------------------------------------------------
extern "C" void kernel_launch(void* const* d_in, const int* in_sizes, int n_in,
                              void* d_out, int out_size, void* d_ws, size_t ws_size,
                              hipStream_t stream)
{
    (void)in_sizes; (void)n_in; (void)out_size; (void)ws_size;
    const float* x    = (const float*)d_in[0];
    const float* w1   = (const float*)d_in[1];
    const float* b1   = (const float*)d_in[2];
    const float* w2   = (const float*)d_in[3];
    const float* b2   = (const float*)d_in[4];
    const float* wfcf = (const float*)d_in[5];
    const float* bfc  = (const float*)d_in[6];
    float* out = (float*)d_out;

    // workspace layout (all 16B aligned)
    char* ws = (char*)d_ws;
    const size_t OFF_H1   = 0;                                 // 2048*676*32*2   = 88,604,672
    const size_t OFF_C2   = OFF_H1 + 88604672ULL;              // 2048*576*64*2   = 150,994,944
    const size_t OFF_POOL = OFF_C2 + 150994944ULL;             // 2048*9216*2     = 37,748,736
    const size_t OFF_WFC  = OFF_POOL + 37748736ULL;            // 32*9216*2       = 589,824
    const size_t OFF_FCO  = OFF_WFC + 589824ULL;               // 2048*32*4       = 262,144
    unsigned short* h1     = (unsigned short*)(ws + OFF_H1);
    unsigned short* c2o    = (unsigned short*)(ws + OFF_C2);
    unsigned short* pooled = (unsigned short*)(ws + OFF_POOL);
    unsigned short* wfcb   = (unsigned short*)(ws + OFF_WFC);
    float*          fcout  = (float*)(ws + OFF_FCO);

    conv1_kernel     <<<5408,  256, 0, stream>>>(x, w1, b1, h1);       // 2048*676 threads
    convert_w_kernel <<<1152,  256, 0, stream>>>(wfcf, wfcb);          // 294912 elems
    conv2_wmma_kernel<<<9216,  256, 0, stream>>>(h1, w2, b2, c2o);     // 1,179,648 rows /128
    pool_kernel      <<<73728, 256, 0, stream>>>(c2o, pooled);         // 18,874,368 elems
    fc_wmma_kernel   <<<64,    256, 0, stream>>>(pooled, wfcb, bfc, fcout); // 2048 rows /32
    quantum_kernel   <<<256,   256, 0, stream>>>(fcout, out);          // 2048 waves
}
